// FusedInt8Linear_56899726737622
// MI455X (gfx1250) — 8-bit matrix operand order repaired and checked by reading, compiled, not run on hardware
//
#include <hip/hip_runtime.h>
#include <stdint.h>

typedef int v8i __attribute__((ext_vector_type(8)));
typedef int v4i __attribute__((ext_vector_type(4)));
typedef int v2i __attribute__((ext_vector_type(2)));

#define MM 16
#define KK 4096
#define NN 16384

__global__ __launch_bounds__(256)
void quant_rows_kernel(const float* __restrict__ x,
                       int8_t* __restrict__ xq,
                       float* __restrict__ xscale) {
    __shared__ float red[256];
    const int row = blockIdx.x;
    const int tid = threadIdx.x;
    const float* xr = x + row * KK;

    float amax = 0.0f;
    #pragma unroll 4
    for (int k = tid; k < KK; k += 256)
        amax = fmaxf(amax, fabsf(xr[k]));
    red[tid] = amax;
    __syncthreads();
    #pragma unroll
    for (int s = 128; s > 0; s >>= 1) {
        if (tid < s) red[tid] = fmaxf(red[tid], red[tid + s]);
        __syncthreads();
    }
    const float scale = fmaxf(red[0], 1e-12f) / 127.0f;
    const float inv   = 1.0f / scale;
    if (tid == 0) xscale[row] = scale;

    int8_t* xqr = xq + row * KK;
    #pragma unroll 4
    for (int k = tid; k < KK; k += 256) {
        float v = rintf(xr[k] * inv);
        v = fminf(fmaxf(v, -128.0f), 127.0f);
        xqr[k] = (int8_t)(int)v;
    }
}

__global__ __launch_bounds__(256, 2)
void gemm_int8_kernel(const int8_t* __restrict__ xq,
                      const int8_t* __restrict__ w,
                      const float* __restrict__ xscale,
                      const float* __restrict__ wscale,
                      const float* __restrict__ bias,
                      float* __restrict__ out) {
    __shared__ int8_t lA[MM * KK];

    const int tid = threadIdx.x;

    {
        const v4i* src = (const v4i*)xq;
        v4i*       dst = (v4i*)lA;
        #pragma unroll
        for (int i = 0; i < 16; ++i)
            dst[tid + i * 256] = src[tid + i * 256];
    }
    __syncthreads();

    const int wave = tid >> 5;
    const int lane = tid & 31;
    const int lrow = lane & 15;
    const int hi   = lane >> 4;

    const int n0 = (blockIdx.x * 8 + wave) * 16;

    const int8_t* wrow = w + (size_t)(n0 + lrow) * KK + (hi << 4);

    const int8_t* arow = lA + lrow * KK + (hi << 4);

    v8i acc = {};

    #pragma unroll 4
    for (int k0 = 0; k0 < KK; k0 += 64) {
        __builtin_prefetch(wrow + k0 + 1024, 0, 1);

        const v4i* bp = (const v4i*)(wrow + k0);
        v4i blo = bp[0];
        v4i bhi = bp[2];

        const v2i* ap = (const v2i*)(arow + k0);
        v2i a0 = ap[0];
        v2i a1 = ap[1];
        v2i a2 = ap[4];
        v2i a3 = ap[5];

        v8i a = { a0[0], a0[1], a1[0], a1[1], a2[0], a2[1], a3[0], a3[1] };
        v8i b = { blo[0], blo[1], blo[2], blo[3], bhi[0], bhi[1], bhi[2], bhi[3] };

        acc = __builtin_amdgcn_wmma_i32_16x16x64_iu8(
                   true, a,  true, b, acc,
                   false,  false);
    }

    const int   n  = n0 + lrow;
    const float ws = wscale[n];
    const float bi = bias[n];
    const int   mbase = hi << 3;
    #pragma unroll
    for (int j = 0; j < 8; ++j) {
        const int m = mbase + j;
        out[(size_t)m * NN + n] = (float)acc[j] * xscale[m] * ws + bi;
    }
}

extern "C" void kernel_launch(void* const* d_in, const int* in_sizes, int n_in,
                              void* d_out, int out_size, void* d_ws, size_t ws_size,
                              hipStream_t stream) {
    const float*  x      = (const float*)d_in[0];
    const int8_t* w      = (const int8_t*)d_in[1];
    const float*  wscale = (const float*)d_in[2];
    const float*  bias   = (const float*)d_in[3];
    float*        out    = (float*)d_out;

    int8_t* xq     = (int8_t*)d_ws;
    float*  xscale = (float*)((char*)d_ws + MM * KK);

    quant_rows_kernel<<<MM, 256, 0, stream>>>(x, xq, xscale);

    gemm_int8_kernel<<<NN / (16 * 8), 256, 0, stream>>>(
        xq, w, xscale, wscale, bias, out);
}
